// GraphSage_89781996355909
// MI455X (gfx1250) — matrix operand order repaired and checked by reading, not compiled, not run on hardware
//
#include <hip/hip_runtime.h>
#include <hip/hip_bf16.h>
#include <math.h>
#include <stdint.h>

// ---------------------------------------------------------------------------
// Types for CDNA5 WMMA
// ---------------------------------------------------------------------------
typedef __attribute__((ext_vector_type(16))) __bf16        v16bf;
typedef __attribute__((ext_vector_type(2)))  __bf16        v2bf;
typedef __attribute__((ext_vector_type(8)))  float         v8f;
typedef __attribute__((ext_vector_type(4)))  unsigned int  v4u;

__device__ __forceinline__ unsigned short f2bf(float f) {
    union { float f; unsigned int u; } x; x.f = f;
    unsigned int u = x.u;
    unsigned int r = u + 0x7FFFu + ((u >> 16) & 1u);   // round-to-nearest-even
    return (unsigned short)(r >> 16);
}

// pack two f32 -> packed bf16 via native fptrunc (RNE); lets the backend pick
// the hardware cvt on gfx1250
__device__ __forceinline__ unsigned int pk2bf(float x, float y) {
    union { v2bf v; unsigned int u; } c;
    c.v[0] = (__bf16)x;
    c.v[1] = (__bf16)y;
    return c.u;
}

// ---- CDNA5 async global->LDS copy (ASYNCcnt-tracked) ----------------------
__device__ __forceinline__ void async_b128(const void* g, void* lds) {
    unsigned loff = (unsigned)(uintptr_t)lds;              // LDS byte offset (low 32 bits)
    unsigned long long ga = (unsigned long long)(uintptr_t)g;
    asm volatile("global_load_async_to_lds_b128 %0, %1, off"
                 :: "v"(loff), "v"(ga) : "memory");
}
__device__ __forceinline__ void wait_asynccnt0() {
    asm volatile("s_wait_asynccnt 0" ::: "memory");
}

// ---------------------------------------------------------------------------
// WMMA GEMM, double-buffered, async-LDS B path, optional split-K:
//   C (+ z partial) = act( A[M,K] * B[K,N] )
//   A : f32 row-major (lda), converted to bf16 on the fly (register path)
//   BT: bf16 TRANSPOSED [N][K] (ldbt), copied global->LDS asynchronously
//   C : f32 row-major (ldc); blockIdx.z selects K-chunk, C += z*cz
// Block: 256 threads (8 wave32), tile 64(M) x 128(N), K-step 32.
// ---------------------------------------------------------------------------
#define TM 64
#define TN 128
#define TK 32
#define LP 40   // LDS row stride (shorts): 80B, 16B-aligned, bank-permuting

__global__ __launch_bounds__(256)
void gemm_bf16_wmma(const float* __restrict__ A, int lda,
                    const unsigned short* __restrict__ BT, int ldbt,
                    float* __restrict__ C, int ldc, size_t cz,
                    int klen, int relu)
{
    __shared__ unsigned short lA[2][TM][LP];   // [buf][m][k]
    __shared__ unsigned short lB[2][TN][LP];   // [buf][n][k]

    const int m0 = blockIdx.x * TM;
    const int n0 = blockIdx.y * TN;
    const int kbase = blockIdx.z * klen;
    C += (size_t)blockIdx.z * cz;

    const int t     = threadIdx.x & 255;
    const int lane  = t & 31;
    const int wave  = t >> 5;
    const int waveM = wave >> 2;               // 0..1
    const int waveN = wave & 3;                // 0..3
    const int mrow  = lane & 15;
    const int hi    = lane >> 4;

    // fill assignments (2 chunks each, no guards)
    const int am = t >> 3, ak = (t & 7) * 4;   // A: rows am, am+32
    const int bn = t >> 2, bk = (t & 3) * 8;   // B: rows bn, bn+64

    const float*          Abase = A  + (size_t)(m0 + am) * lda  + kbase + ak;
    const unsigned short* Bbase = BT + (size_t)(n0 + bn) * ldbt + kbase + bk;
    const size_t AstepM = (size_t)32 * lda;
    const size_t BstepN = (size_t)64 * ldbt;

    float4 a0, a1;
    auto loadA = [&](int k0) {
        a0 = *(const float4*)(Abase + k0);
        a1 = *(const float4*)(Abase + k0 + AstepM);
    };
    auto storeA = [&](int buf) {
        *(uint2*)&lA[buf][am][ak]      = make_uint2(pk2bf(a0.x, a0.y), pk2bf(a0.z, a0.w));
        *(uint2*)&lA[buf][am + 32][ak] = make_uint2(pk2bf(a1.x, a1.y), pk2bf(a1.z, a1.w));
    };
    auto asyncB = [&](int k0, int buf) {       // global -> LDS, no VGPR round-trip
        async_b128(Bbase + k0,          &lB[buf][bn][bk]);
        async_b128(Bbase + k0 + BstepN, &lB[buf][bn + 64][bk]);
    };

    v8f acc[2][2] = {};

    // prologue: stage tile 0
    loadA(0);
    asyncB(0, 0);
    storeA(0);
    wait_asynccnt0();
    __syncthreads();

    int buf = 0;
    for (int k0 = 0; k0 < klen; k0 += TK) {
        const int nxt = k0 + TK;
        if (nxt < klen) {
            loadA(nxt);                 // A regs for next tile
            asyncB(nxt, buf ^ 1);       // B directly into the other LDS buffer
        }
        if (nxt + TK < klen)
            __builtin_prefetch(Abase + nxt + TK, 0, 1);   // global_prefetch_b8

        // ---- fragments from current buffer (contiguous 16B chunks) ----
        union Frag { v16bf v; v4u q[2]; };
        Frag afr[2], bfr[2];
        #pragma unroll
        for (int mi = 0; mi < 2; ++mi) {
            int m = waveM * 32 + mi * 16 + mrow;
            afr[mi].q[0] = *(const v4u*)&lA[buf][m][hi * 8];
            afr[mi].q[1] = *(const v4u*)&lA[buf][m][16 + hi * 8];
        }
        #pragma unroll
        for (int ni = 0; ni < 2; ++ni) {
            int n = waveN * 32 + ni * 16 + mrow;
            bfr[ni].q[0] = *(const v4u*)&lB[buf][n][hi * 8];
            bfr[ni].q[1] = *(const v4u*)&lB[buf][n][16 + hi * 8];
        }

        // ---- 4x v_wmma_f32_16x16x32_bf16 ----
        #pragma unroll
        for (int mi = 0; mi < 2; ++mi)
            #pragma unroll
            for (int ni = 0; ni < 2; ++ni)
                acc[mi][ni] = __builtin_amdgcn_wmma_f32_16x16x32_bf16(
                    false, afr[mi].v, false, bfr[ni].v,
                    (short)0, acc[mi][ni], false, false);

        __syncthreads();                 // all waves done reading lds[buf]
        if (nxt < klen) storeA(buf ^ 1);
        wait_asynccnt0();                // B tile landed in lds[buf^1]
        __syncthreads();                 // publish next buffer
        buf ^= 1;
    }

    // ---- store C (D layout: lane -> N, VGPR v -> M = v + 8*hi) ----
    #pragma unroll
    for (int mi = 0; mi < 2; ++mi)
        #pragma unroll
        for (int ni = 0; ni < 2; ++ni) {
            int n = n0 + waveN * 32 + ni * 16 + mrow;
            #pragma unroll
            for (int v = 0; v < 8; ++v) {
                int m = m0 + waveM * 32 + mi * 16 + v + 8 * hi;
                float x = acc[mi][ni][v];
                if (relu) x = fmaxf(x, 0.0f);
                C[(size_t)m * ldc + n] = x;
            }
        }
}

// ---------------------------------------------------------------------------
// Split-K reduction: C[i/N*ldc + i%N] = sum_z P[z*zstride + i]
// ---------------------------------------------------------------------------
__global__ void reduce_partials(const float* __restrict__ P, int parts, size_t zstride,
                                float* __restrict__ C, int ldc, int N, int total)
{
    int i = blockIdx.x * blockDim.x + threadIdx.x;
    if (i >= total) return;
    float s = 0.0f;
    for (int z = 0; z < parts; ++z) s += P[(size_t)z * zstride + i];
    C[(size_t)(i / N) * ldc + (i % N)] = s;
}

// ---------------------------------------------------------------------------
// Helpers
// ---------------------------------------------------------------------------
__global__ void compose_idx(const int* __restrict__ a, const int* __restrict__ b,
                            int* __restrict__ out, int n)
{
    int i = blockIdx.x * blockDim.x + threadIdx.x;
    if (i < n) out[i] = a[b[i]];
}

__global__ void gather_rows(const float* __restrict__ src, int lds,
                            const int* __restrict__ idx,
                            float* __restrict__ out, int ldo, int cols)
{
    int r = blockIdx.x;
    int row = idx[r];
    for (int c = threadIdx.x; c < cols; c += blockDim.x)
        out[(size_t)r * ldo + c] = src[(size_t)row * lds + c];
}

// BT[c][r] = bf16( src[ idx ? idx[r] : r ][c] )  -- gather + convert + transpose
__global__ void gather_convert_T(const float* __restrict__ src, int lds,
                                 const int* __restrict__ idx,
                                 unsigned short* __restrict__ BT,
                                 int rows, int cols)
{
    int r = blockIdx.x;
    int srow = idx ? idx[r] : r;
    for (int c = threadIdx.x; c < cols; c += blockDim.x)
        BT[(size_t)c * rows + r] = f2bf(src[(size_t)srow * lds + c]);
}

// ---------------------------------------------------------------------------
// Final: per-row l2-normalize + 5-layer MLP + single-unit softmax
// ---------------------------------------------------------------------------
__global__ __launch_bounds__(64)
void final_mlp(const float* __restrict__ OUT,
               const float* __restrict__ W1, const float* __restrict__ b1,
               const float* __restrict__ W2, const float* __restrict__ b2,
               const float* __restrict__ W3, const float* __restrict__ b3,
               const float* __restrict__ W4, const float* __restrict__ b4,
               const float* __restrict__ W5, const float* __restrict__ b5,
               float* __restrict__ out)
{
    __shared__ float e[256];
    __shared__ float h[64];
    __shared__ float red[64];

    const int r = blockIdx.x;
    const int t = threadIdx.x & 63;

    float v0 = OUT[(size_t)r * 256 + t];
    float v1 = OUT[(size_t)r * 256 + 64 + t];
    float v2 = OUT[(size_t)r * 256 + 128 + t];
    float v3 = OUT[(size_t)r * 256 + 192 + t];
    red[t] = v0 * v0 + v1 * v1 + v2 * v2 + v3 * v3;
    __syncthreads();
    for (int s = 32; s > 0; s >>= 1) {
        if (t < s) red[t] += red[t + s];
        __syncthreads();
    }
    float scale = rsqrtf(fmaxf(red[0], 1e-12f));
    e[t] = v0 * scale; e[64 + t] = v1 * scale;
    e[128 + t] = v2 * scale; e[192 + t] = v3 * scale;
    __syncthreads();

    {   // L1: 256 -> 64, relu
        float a = b1[t];
        for (int c = 0; c < 256; ++c) a += e[c] * W1[c * 64 + t];
        h[t] = fmaxf(a, 0.0f);
    }
    __syncthreads();
    float h2 = 0.0f;   // L2: 64 -> 32, relu
    if (t < 32) { float a = b2[t]; for (int c = 0; c < 64; ++c) a += h[c] * W2[c * 32 + t]; h2 = fmaxf(a, 0.0f); }
    __syncthreads();
    if (t < 32) h[t] = h2;
    __syncthreads();
    float h3 = 0.0f;   // L3: 32 -> 16, relu
    if (t < 16) { float a = b3[t]; for (int c = 0; c < 32; ++c) a += h[c] * W3[c * 16 + t]; h3 = fmaxf(a, 0.0f); }
    __syncthreads();
    if (t < 16) h[t] = h3;
    __syncthreads();
    float h4 = 0.0f;   // L4: 16 -> 8, relu
    if (t < 8) { float a = b4[t]; for (int c = 0; c < 16; ++c) a += h[c] * W4[c * 8 + t]; h4 = fmaxf(a, 0.0f); }
    __syncthreads();
    if (t < 8) h[t] = h4;
    __syncthreads();
    if (t == 0) {      // L5: 8 -> 1, softmax over a single logit == 1.0
        float z = b5[0];
        for (int c = 0; c < 8; ++c) z += h[c] * W5[c];
        float ez = __expf(z - z);
        out[r] = ez / ez;
    }
}

// ---------------------------------------------------------------------------
// Launch
// ---------------------------------------------------------------------------
extern "C" void kernel_launch(void* const* d_in, const int* in_sizes, int n_in,
                              void* d_out, int out_size, void* d_ws, size_t ws_size,
                              hipStream_t stream)
{
    (void)in_sizes; (void)n_in; (void)out_size; (void)ws_size;

    const int N2 = 16384, N1 = 2048, Bsz = 512, FEAT = 256, INT = 128;
    const int KS = 4;   // split-K factor for the two gather-GEMMs

    const float* feats  = (const float*)d_in[0];
    const float* w_agg1 = (const float*)d_in[1];   // [512,128]
    const float* w_agg2 = (const float*)d_in[2];   // [256,128]
    const float* W1 = (const float*)d_in[3];  const float* b1 = (const float*)d_in[4];
    const float* W2 = (const float*)d_in[5];  const float* b2 = (const float*)d_in[6];
    const float* W3 = (const float*)d_in[7];  const float* b3 = (const float*)d_in[8];
    const float* W4 = (const float*)d_in[9];  const float* b4 = (const float*)d_in[10];
    const float* W5 = (const float*)d_in[11]; const float* b5 = (const float*)d_in[12];

    struct Branch {
        const int* nodes; const int* s2s2; const int* s2d2; const float* dif2;
        const int* s2s1;  const int* s2d1; const float* dif1;
    };
    Branch br[2] = {
        { (const int*)d_in[13], (const int*)d_in[14], (const int*)d_in[15],
          (const float*)d_in[16], (const int*)d_in[17], (const int*)d_in[18],
          (const float*)d_in[19] },
        { (const int*)d_in[20], (const int*)d_in[21], (const int*)d_in[22],
          (const float*)d_in[23], (const int*)d_in[24], (const int*)d_in[25],
          (const float*)d_in[26] },
    };

    // ---- workspace carve-up (~38 MB) ----
    char* p = (char*)d_ws;
    auto carve = [&](size_t bytes) {
        void* r = (void*)p;
        p += (bytes + 255) & ~(size_t)255;
        return r;
    };
    float* X1[2];  float* H1[2];  float* X2[2];
    unsigned short* srcBT[2]; unsigned short* H1T[2];
    int* bidx2[2]; int* didx2[2];
    for (int b = 0; b < 2; ++b) {
        X1[b]    = (float*)carve((size_t)N1 * 512 * sizeof(float));
        H1[b]    = (float*)carve((size_t)N1 * INT * sizeof(float));
        X2[b]    = (float*)carve((size_t)Bsz * 256 * sizeof(float));
        srcBT[b] = (unsigned short*)carve((size_t)FEAT * N2 * sizeof(short));
        H1T[b]   = (unsigned short*)carve((size_t)INT * N1 * sizeof(short));
        bidx2[b] = (int*)carve((size_t)N2 * sizeof(int));
        didx2[b] = (int*)carve((size_t)N1 * sizeof(int));
    }
    float* OUTC = (float*)carve((size_t)Bsz * 256 * sizeof(float));
    unsigned short* wagg1T = (unsigned short*)carve((size_t)INT * 512 * sizeof(short));
    unsigned short* wagg2T = (unsigned short*)carve((size_t)INT * 256 * sizeof(short));
    float* Cp1 = (float*)carve((size_t)KS * N1 * FEAT * sizeof(float));  // 8 MB
    float* Cp2 = (float*)carve((size_t)KS * Bsz * INT * sizeof(float));  // 1 MB

    // ---- weight transposes (bf16, once) ----
    gather_convert_T<<<512, 128, 0, stream>>>(w_agg1, INT, nullptr, wagg1T, 512, INT);
    gather_convert_T<<<256, 128, 0, stream>>>(w_agg2, INT, nullptr, wagg2T, 256, INT);

    for (int b = 0; b < 2; ++b) {
        compose_idx<<<(N2 + 255) / 256, 256, 0, stream>>>(br[b].nodes, br[b].s2s2, bidx2[b], N2);
        compose_idx<<<(N1 + 255) / 256, 256, 0, stream>>>(br[b].nodes, br[b].s2d2, didx2[b], N1);

        // srcBT = bf16(feats[nodes[s2s2]])^T   [256][16384]
        gather_convert_T<<<N2, 256, 0, stream>>>(feats, FEAT, bidx2[b], srcBT[b], N2, FEAT);

        // X1 right half = dst2 = feats[nodes[s2d2]]
        gather_rows<<<N1, 256, 0, stream>>>(feats, FEAT, didx2[b], X1[b] + FEAT, 512, FEAT);

        // agg2 = dif2 @ src2  [2048,16384]x[16384,256], split-K x4 -> 256 blocks
        gemm_bf16_wmma<<<dim3(N1 / TM, FEAT / TN, KS), 256, 0, stream>>>(
            br[b].dif2, N2, srcBT[b], N2, Cp1, FEAT, (size_t)N1 * FEAT, N2 / KS, 0);
        reduce_partials<<<(N1 * FEAT + 255) / 256, 256, 0, stream>>>(
            Cp1, KS, (size_t)N1 * FEAT, X1[b], 512, FEAT, N1 * FEAT);

        // H1 = relu(X1 @ w_agg1)   [2048,512]x[512,128]
        gemm_bf16_wmma<<<dim3(N1 / TM, INT / TN, 1), 256, 0, stream>>>(
            X1[b], 512, wagg1T, 512, H1[b], INT, 0, 512, 1);

        // X2 right half = dst1 = H1[s2d1]
        gather_rows<<<Bsz, 128, 0, stream>>>(H1[b], INT, br[b].s2d1, X2[b] + INT, 256, INT);

        // H1T = bf16(H1[s2s1])^T   [128][2048]
        gather_convert_T<<<N1, 128, 0, stream>>>(H1[b], INT, br[b].s2s1, H1T[b], N1, INT);

        // agg1 = dif1 @ H1[s2s1]  [512,2048]x[2048,128], split-K x4 -> 32 blocks
        gemm_bf16_wmma<<<dim3(Bsz / TM, INT / TN, KS), 256, 0, stream>>>(
            br[b].dif1, N1, H1T[b], N1, Cp2, INT, (size_t)Bsz * INT, N1 / KS, 0);
        reduce_partials<<<(Bsz * INT + 255) / 256, 256, 0, stream>>>(
            Cp2, KS, (size_t)Bsz * INT, X2[b], 256, INT, Bsz * INT);

        // OUTC cols [128b,128b+128) = X2 @ w_agg2   [512,256]x[256,128]
        gemm_bf16_wmma<<<dim3(Bsz / TM, INT / TN, 1), 256, 0, stream>>>(
            X2[b], 256, wagg2T, 256, OUTC + (size_t)INT * b, 256, 0, 256, 0);
    }

    final_mlp<<<Bsz, 64, 0, stream>>>(OUTC, W1, b1, W2, b2, W3, b3, W4, b4, W5, b5,
                                      (float*)d_out);
}
